// MixOfMambaModule_11158325035380
// MI455X (gfx1250) — matrix operand order repaired and checked by reading, compiled, not run on hardware
//
#include <hip/hip_runtime.h>
#include <hip/hip_bf16.h>

// ---------------------------------------------------------------------------
// MixOfMamba on MI455X (gfx1250): dense GEMMs via v_wmma_f32_16x16x32_f16 with
// 2x2 register blocking; selective scan with TDM (tensor_load_to_lds) double
// buffering of the per-step B_t/C_t vectors into LDS.
// ---------------------------------------------------------------------------

typedef __attribute__((ext_vector_type(16))) _Float16 v16h;
typedef __attribute__((ext_vector_type(8)))  float    v8f;
typedef unsigned int uint32x4 __attribute__((ext_vector_type(4)));
typedef int          int32x4  __attribute__((ext_vector_type(4)));
typedef int          int32x8  __attribute__((ext_vector_type(8)));

union HFrag { uint4 q[2]; v16h v; };

#define Bn 2
#define Cc 128
#define LL 4096            // H*W
#define EE 3
#define DIN 256
#define DSTATE 64
#define DTR 8
#define CONVDIM 64
#define MROWS 8192         // B*L

#if __has_builtin(__builtin_amdgcn_tensor_load_to_lds) && __has_builtin(__builtin_amdgcn_s_wait_tensorcnt)
#define USE_TDM 1
#else
#define USE_TDM 0
#endif

// Compile-time probe feedback (shows up in hipcc stderr):
#define TDM_STR2(x) #x
#define TDM_STR(x) TDM_STR2(x)
#if USE_TDM
#pragma message("CDNA5-PROBE: USE_TDM=1 (tensor_load_to_lds + s_wait_tensorcnt builtins available), clang=" TDM_STR(__clang_major__))
#else
#pragma message("CDNA5-PROBE: USE_TDM=0 (TDM builtins unavailable, scalar LDS fill fallback), clang=" TDM_STR(__clang_major__))
#endif

#if USE_TDM
// 1-D TDM descriptor: copy nElems f32 from gsrc (global) to ldsOff (LDS bytes).
__device__ __forceinline__ void tdm_load_row(const float* gsrc, unsigned ldsOff, int nElems)
{
    unsigned long long ga = (unsigned long long)(size_t)gsrc;
    uint32x4 g0;
    g0.x = 1u;                                            // count=1, user descriptor
    g0.y = ldsOff;                                        // lds_addr [63:32]
    g0.z = (unsigned)(ga & 0xFFFFFFFFu);                  // global_addr [95:64]
    g0.w = (unsigned)((ga >> 32) & 0x01FFFFFFu) | (2u << 30);  // addr[56:32] | type=2
    int32x8 g1;
    g1[0] = 0x20000;                                      // data_size=2 (4B), wg_mask=0
    g1[1] = nElems << 16;                                 // tensor_dim0[15:0] @ bits 48+
    g1[2] = 1 << 16;                                      // tensor_dim1 = 1
    g1[3] = nElems << 16;                                 // tile_dim0 @ bits 112+
    g1[4] = 0;                                            // tile_dim1/2 unused (1-D)
    g1[5] = nElems;                                       // tensor_dim0_stride low bits
    g1[6] = 0;
    g1[7] = 0;
    int32x4 z4 = {0, 0, 0, 0};
#if defined(__clang_major__) && (__clang_major__ >= 23)
    int32x8 z8 = {0, 0, 0, 0, 0, 0, 0, 0};
    __builtin_amdgcn_tensor_load_to_lds(g0, g1, z4, z4, z8, 0);
#else
    __builtin_amdgcn_tensor_load_to_lds(g0, g1, z4, z4, 0);
#endif
}
#endif

// ---------------------------------------------------------------------------
// WMMA GEMM with 2x2 tile blocking:  D(M,N) = A(M,K) * W(N,K)^T, f16 in f32 out.
// One wave computes a 32x32 output block = 4 WMMA tiles; 4 WMMAs per K-step of 32.
// ---------------------------------------------------------------------------
__global__ void gemm_wmma_f16(const _Float16* __restrict__ A, long strideAe,
                              const _Float16* __restrict__ W, long strideWe,
                              float* __restrict__ D, long strideDe,
                              int M, int Ntiles, int Nvalid, int K,
                              int lda, int ldw, int ldd)
{
    int e = blockIdx.y;
    A += (long)e * strideAe;
    W += (long)e * strideWe;
    D += (long)e * strideDe;

    int wavesPerBlock = blockDim.x >> 5;
    int st = blockIdx.x * wavesPerBlock + (threadIdx.x >> 5);
    int Mst = M >> 5;                  // 32-row super-tiles
    int Nst = (Ntiles + 1) >> 1;       // 32-col super-tiles
    if (st >= Mst * Nst) return;
    int mt = st / Nst;
    int nt = st % Nst;

    int lane = threadIdx.x & 31;
    int half = lane >> 4;              // 0: lanes 0-15, 1: lanes 16-31
    int l16  = lane & 15;

    int arow0 = (mt << 5) + l16;
    int brow0 = (nt << 5) + l16;
    int brow1 = brow0 + 16;
    if (brow0 >= Nvalid) brow0 = Nvalid - 1;   // clamp OOB weight rows (stores guarded)
    if (brow1 >= Nvalid) brow1 = Nvalid - 1;

    // A-fragment lane view: row, K = {k0+8h..+7} U {k0+8h+16..+23}
    const _Float16* A0 = A + (long)arow0 * lda + (half << 3);
    const _Float16* A1 = A0 + (long)16 * lda;
    // B-fragment lane view (B = W^T): col, K = {k0+8h..+7} U {k0+8h+16..+23}
    const _Float16* B0 = W + (long)brow0 * ldw + (half << 3);
    const _Float16* B1 = W + (long)brow1 * ldw + (half << 3);

    v8f acc00 = {}, acc01 = {}, acc10 = {}, acc11 = {};
    for (int k0 = 0; k0 < K; k0 += 32) {
        HFrag a0, a1, b0, b1;
        a0.q[0] = *(const uint4*)(A0 + k0);  a0.q[1] = *(const uint4*)(A0 + k0 + 16);
        a1.q[0] = *(const uint4*)(A1 + k0);  a1.q[1] = *(const uint4*)(A1 + k0 + 16);
        b0.q[0] = *(const uint4*)(B0 + k0);  b0.q[1] = *(const uint4*)(B0 + k0 + 16);
        b1.q[0] = *(const uint4*)(B1 + k0);  b1.q[1] = *(const uint4*)(B1 + k0 + 16);
        if (k0 + 32 < K) {
            __builtin_prefetch(A0 + k0 + 32, 0, 1);   // global_prefetch_b8
            __builtin_prefetch(A1 + k0 + 32, 0, 1);
            __builtin_prefetch(B0 + k0 + 32, 0, 1);
            __builtin_prefetch(B1 + k0 + 32, 0, 1);
        }
        acc00 = __builtin_amdgcn_wmma_f32_16x16x32_f16(false, a0.v, false, b0.v, (short)0, acc00, false, false);
        acc01 = __builtin_amdgcn_wmma_f32_16x16x32_f16(false, a0.v, false, b1.v, (short)0, acc01, false, false);
        acc10 = __builtin_amdgcn_wmma_f32_16x16x32_f16(false, a1.v, false, b0.v, (short)0, acc10, false, false);
        acc11 = __builtin_amdgcn_wmma_f32_16x16x32_f16(false, a1.v, false, b1.v, (short)0, acc11, false, false);
    }

    int col0 = (nt << 5) + l16;
    int col1 = col0 + 16;
    int r0   = (mt << 5) + (half << 3);
    #pragma unroll
    for (int r = 0; r < 8; ++r) {
        if (col0 < Nvalid) {
            D[(long)(r0 + r) * ldd + col0]      = acc00[r];
            D[(long)(r0 + 16 + r) * ldd + col0] = acc10[r];
        }
        if (col1 < Nvalid) {
            D[(long)(r0 + r) * ldd + col1]      = acc01[r];
            D[(long)(r0 + 16 + r) * ldd + col1] = acc11[r];
        }
    }
}

// ---------------------------------------------------------------------------
// Fusion GEMM: stores transposed into (B, C, H*W) + bias.
// ---------------------------------------------------------------------------
__global__ void gemm_wmma_fusion(const _Float16* __restrict__ A,
                                 const _Float16* __restrict__ W,
                                 const float* __restrict__ bias,
                                 float* __restrict__ out,
                                 int Ntiles)
{
    int wavesPerBlock = blockDim.x >> 5;
    int tile = blockIdx.x * wavesPerBlock + (threadIdx.x >> 5);
    if (tile >= (MROWS >> 4) * Ntiles) return;
    int mt = tile / Ntiles;
    int nt = tile % Ntiles;

    int lane = threadIdx.x & 31;
    int half = lane >> 4;
    int l16  = lane & 15;

    const _Float16* Aptr = A + (long)((mt << 4) + l16) * Cc + (half << 3);
    const _Float16* Wptr = W + (long)((nt << 4) + l16) * Cc + (half << 3);

    v8f acc = {};
    for (int k0 = 0; k0 < Cc; k0 += 32) {
        HFrag af, bf;
        af.q[0] = *(const uint4*)(Aptr + k0);
        af.q[1] = *(const uint4*)(Aptr + k0 + 16);
        bf.q[0] = *(const uint4*)(Wptr + k0);
        bf.q[1] = *(const uint4*)(Wptr + k0 + 16);
        acc = __builtin_amdgcn_wmma_f32_16x16x32_f16(false, af.v, false, bf.v, (short)0, acc, false, false);
    }

    int col = (nt << 4) + l16;
    float bv = bias[col];
    int rbase = (mt << 4) + (half << 3);
    #pragma unroll
    for (int r = 0; r < 8; ++r) {
        int row = rbase + r;            // row = b*4096 + hw
        int bb  = row >> 12;
        int hw  = row & 4095;
        out[((long)bb * Cc + col) * LL + hw] = acc[r] + bv;
    }
}

// ---------------------------------------------------------------------------
// Elementwise helpers
// ---------------------------------------------------------------------------
__global__ void cast_f32_to_f16(const float* __restrict__ s, _Float16* __restrict__ d, int n) {
    int i = blockIdx.x * blockDim.x + threadIdx.x;
    if (i < n) d[i] = (_Float16)s[i];
}

// x (B,C,H,W) f32 -> xseq (B, HW, C) f16
__global__ void make_xseq_f16(const float* __restrict__ x, _Float16* __restrict__ d) {
    int i = blockIdx.x * blockDim.x + threadIdx.x;
    if (i >= Bn * LL * Cc) return;
    int c  = i & 127;
    int hw = (i >> 7) & 4095;
    int b  = i >> 19;
    d[i] = (_Float16)x[(((long)b * Cc + c) << 12) + hw];
}

// Router: 3x3 SAME conv (C=128 -> 64) + ReLU + spatial mean. One block per (b,oc).
__global__ void router_conv_pool(const float* __restrict__ x, const float* __restrict__ w,
                                 const float* __restrict__ cb, float* __restrict__ pooled)
{
    int b  = blockIdx.x >> 6;
    int oc = blockIdx.x & 63;
    int t  = threadIdx.x;
    float bias = cb[oc];
    float sum = 0.f;
    for (int p = t; p < LL; p += 256) {
        int py = p >> 6, px = p & 63;
        float acc = bias;
        for (int ic = 0; ic < Cc; ++ic) {
            const float* xp = x + ((((long)b << 7) + ic) << 12);
            const float* wp = w + ((long)(oc << 7) + ic) * 9;
            #pragma unroll
            for (int kh = 0; kh < 3; ++kh) {
                int yy = py + kh - 1;
                if ((unsigned)yy < 64u) {
                    const float* xr = xp + (yy << 6);
                    #pragma unroll
                    for (int kw = 0; kw < 3; ++kw) {
                        int xx = px + kw - 1;
                        if ((unsigned)xx < 64u) acc += wp[kh * 3 + kw] * xr[xx];
                    }
                }
            }
        }
        sum += fmaxf(acc, 0.f);
    }
    __shared__ float red[256];
    red[t] = sum;
    __syncthreads();
    for (int s = 128; s > 0; s >>= 1) { if (t < s) red[t] += red[t + s]; __syncthreads(); }
    if (t == 0) pooled[(b << 6) + oc] = red[0] * (1.f / (float)LL);
}

__global__ void router_softmax(const float* __restrict__ pooled, const float* __restrict__ lw,
                               const float* __restrict__ lb, float* __restrict__ wts)
{
    int b = threadIdx.x;
    if (b >= Bn) return;
    float l[EE];
    for (int e = 0; e < EE; ++e) {
        float z = lb[e];
        for (int c = 0; c < CONVDIM; ++c) z += pooled[(b << 6) + c] * lw[(e << 6) + c];
        l[e] = z;
    }
    float m = fmaxf(l[0], fmaxf(l[1], l[2]));
    float ex[EE], s = 0.f;
    for (int e = 0; e < EE; ++e) { ex[e] = __expf(l[e] - m); s += ex[e]; }
    for (int e = 0; e < EE; ++e) wts[b * EE + e] = ex[e] / s;
}

// Causal depthwise conv1d (k=4) + bias + SiLU on the x-half of XR.
__global__ void dwconv_silu(const float* __restrict__ XR, const float* __restrict__ cw,
                            const float* __restrict__ cb, float* __restrict__ xsf,
                            _Float16* __restrict__ xsh)
{
    int i = blockIdx.x * blockDim.x + threadIdx.x;
    if (i >= EE * MROWS * DIN) return;
    int d  = i & 255;
    int bl = (i >> 8) & 8191;
    int e  = i >> 21;
    int l  = bl & 4095;
    long base = ((long)e << 22) + (long)(bl - l) * (2 * DIN) + d;
    float acc = cb[(e << 8) + d];
    const float* w = cw + (((long)(e << 8) + d) << 2);
    #pragma unroll
    for (int k = 0; k < 4; ++k) {
        int lp = l - 3 + k;
        if (lp >= 0) acc += w[k] * XR[base + (long)lp * (2 * DIN)];
    }
    float s = acc / (1.f + __expf(-acc));   // silu
    xsf[i] = s;
    xsh[i] = (_Float16)s;
}

// delta = softplus(dt @ dt_w^T + dt_b); dt = dbl[..., :8]
__global__ void delta_kernel(const float* __restrict__ dbl, const float* __restrict__ dtw,
                             const float* __restrict__ dtb, float* __restrict__ delta)
{
    int i = blockIdx.x * blockDim.x + threadIdx.x;
    if (i >= EE * MROWS * DIN) return;
    int d  = i & 255;
    int ri = (i >> 8) & 8191;
    int e  = i >> 21;
    const float* dr = dbl + (long)e * MROWS * 136 + (long)ri * 136;
    float z = dtb[(e << 8) + d];
    const float* wv = dtw + (((long)(e << 8) + d) << 3);
    #pragma unroll
    for (int r = 0; r < DTR; ++r) z += dr[r] * wv[r];
    delta[i] = (z > 20.f) ? z : log1pf(__expf(z));
}

// Sequential selective scan. grid = E*B blocks of 256 threads; thread = channel d.
// h[64] and A row live in registers. B_t/C_t (128 contiguous floats at dblrow+8)
// are DMA'd into LDS by the TDM one step ahead (double-buffered), synchronized
// with s_wait_tensorcnt in wave 0 + a workgroup barrier.
__global__ void __launch_bounds__(256, 1)
scan_kernel(const float* __restrict__ dbl, float* __restrict__ deltays,
            const float* __restrict__ xsf, const float* __restrict__ Alog)
{
    int eb = blockIdx.x;
    int e = eb >> 1, b = eb & 1;
    int t = threadIdx.x;
    __shared__ float sBC[2][2 * DSTATE];

    float a[DSTATE], h[DSTATE];
    const float* ap = Alog + (((long)(e << 8) + t) << 6);
    #pragma unroll
    for (int n = 0; n < DSTATE; ++n) { a[n] = -__expf(ap[n]); h[n] = 0.f; }

    const float* dble = dbl + (long)e * MROWS * 136 + (long)b * LL * 136;
    long rowBase = ((long)e * MROWS + (long)b * LL);

#if USE_TDM
    if (t < 32) {
        tdm_load_row(dble + 8, (unsigned)(size_t)(void*)&sBC[0][0], 2 * DSTATE);
    }
    for (int l = 0; l < LL; ++l) {
        __syncthreads();          // all waves done reading buf[(l+1)&1] from step l-1
        if (t < 32) {
            if (l + 1 < LL) {
                tdm_load_row(dble + (long)(l + 1) * 136 + 8,
                             (unsigned)(size_t)(void*)&sBC[(l + 1) & 1][0], 2 * DSTATE);
                __builtin_amdgcn_s_wait_tensorcnt(1);   // step-l load complete (in-order)
            } else {
                __builtin_amdgcn_s_wait_tensorcnt(0);
            }
        }
        __syncthreads();          // publish buf[l&1] to all waves
        const float* sB = &sBC[l & 1][0];
        const float* sC = sB + DSTATE;
#else
    for (int l = 0; l < LL; ++l) {
        const float* dblrow = dble + (long)l * 136;
        __syncthreads();
        if (t < 128) sBC[0][t] = dblrow[8 + t];
        __syncthreads();
        const float* sB = &sBC[0][0];
        const float* sC = sB + DSTATE;
#endif
        long idx = ((rowBase + l) << 8) + t;
        float dlt = deltays[idx];
        float u   = xsf[idx];
        float du  = dlt * u;
        float y = 0.f;
        #pragma unroll
        for (int n = 0; n < DSTATE; ++n) {
            float dA = __expf(dlt * a[n]);
            h[n] = dA * h[n] + du * sB[n];
            y += h[n] * sC[n];
        }
        deltays[idx] = y;       // same slot this thread just consumed
    }
}

// y = (ys + xs*D) * silu(res); res is the second half of XR. Output f16 for GEMM3.
__global__ void ymix_kernel(const float* __restrict__ deltays, const float* __restrict__ xsf,
                            const float* __restrict__ XR, const float* __restrict__ Dp,
                            _Float16* __restrict__ yh)
{
    int i = blockIdx.x * blockDim.x + threadIdx.x;
    if (i >= EE * MROWS * DIN) return;
    int d  = i & 255;
    int ri = (i >> 8) & 8191;
    int e  = i >> 21;
    float res = XR[((long)e << 22) + ((long)ri << 9) + DIN + d];
    float sil = res / (1.f + __expf(-res));
    yh[i] = (_Float16)((deltays[i] + xsf[i] * Dp[(e << 8) + d]) * sil);
}

// Mixture over experts -> (B*HW, C) f16 for the fusion GEMM.
__global__ void combine_kernel(const float* __restrict__ outE, const float* __restrict__ wts,
                               _Float16* __restrict__ wsmix)
{
    int i = blockIdx.x * blockDim.x + threadIdx.x;
    if (i >= Bn * LL * Cc) return;
    int c  = i & 127;
    int hw = (i >> 7) & 4095;
    int b  = i >> 19;
    float acc = 0.f;
    #pragma unroll
    for (int e = 0; e < EE; ++e)
        acc += wts[b * EE + e] * outE[(((long)e * Bn + b) * LL + hw) * Cc + c];
    wsmix[i] = (_Float16)acc;
}

// ---------------------------------------------------------------------------
extern "C" void kernel_launch(void* const* d_in, const int* in_sizes, int n_in,
                              void* d_out, int out_size, void* d_ws, size_t ws_size,
                              hipStream_t stream)
{
    const float* x        = (const float*)d_in[0];
    const float* in_w     = (const float*)d_in[1];
    const float* conv_w   = (const float*)d_in[2];
    const float* conv_b   = (const float*)d_in[3];
    const float* xproj_w  = (const float*)d_in[4];
    const float* dt_w     = (const float*)d_in[5];
    const float* dt_b     = (const float*)d_in[6];
    const float* A_log    = (const float*)d_in[7];
    const float* D_p      = (const float*)d_in[8];
    const float* out_w    = (const float*)d_in[9];
    const float* r_conv_w = (const float*)d_in[10];
    const float* r_conv_b = (const float*)d_in[11];
    const float* r_lin_w  = (const float*)d_in[12];
    const float* r_lin_b  = (const float*)d_in[13];
    const float* fusion_w = (const float*)d_in[14];
    const float* fusion_b = (const float*)d_in[15];
    float* out = (float*)d_out;

    // ---- workspace layout (bytes, 256-aligned) ----
    char* ws = (char*)d_ws;
    size_t o = 0;
    auto take = [&](size_t bytes) { size_t r = o; o += (bytes + 255) & ~(size_t)255; return r; };

    _Float16* xseq_h = (_Float16*)(ws + take((size_t)MROWS * Cc * 2));
    _Float16* inw_h  = (_Float16*)(ws + take((size_t)EE * 2 * DIN * Cc * 2));
    _Float16* xpw_h  = (_Float16*)(ws + take((size_t)EE * 136 * DIN * 2));
    _Float16* outw_h = (_Float16*)(ws + take((size_t)EE * Cc * DIN * 2));
    _Float16* fusw_h = (_Float16*)(ws + take((size_t)Cc * Cc * 2));
    float*    pooled = (float*)   (ws + take((size_t)Bn * CONVDIM * 4));
    float*    wts    = (float*)   (ws + take((size_t)Bn * EE * 4));
    float*    XR     = (float*)   (ws + take((size_t)EE * MROWS * 2 * DIN * 4));
    float*    xsf    = (float*)   (ws + take((size_t)EE * MROWS * DIN * 4));
    _Float16* xsh    = (_Float16*)(ws + take((size_t)EE * MROWS * DIN * 2)); // reused as y_f16
    float*    dbl    = (float*)   (ws + take((size_t)EE * MROWS * 136 * 4));
    float*    delta  = (float*)   (ws + take((size_t)EE * MROWS * DIN * 4)); // reused as ys
    float*    outE   = (float*)   (ws + take((size_t)EE * MROWS * Cc * 4));
    _Float16* wsmix  = (_Float16*)(ws + take((size_t)Bn * LL * Cc * 2));
    (void)ws_size; (void)in_sizes; (void)n_in; (void)out_size;

    const int TPB = 256;
    const int nEW = EE * MROWS * DIN;

    // 1) casts / layout transforms
    make_xseq_f16<<<(Bn * LL * Cc + TPB - 1) / TPB, TPB, 0, stream>>>(x, xseq_h);
    cast_f32_to_f16<<<(EE * 2 * DIN * Cc + TPB - 1) / TPB, TPB, 0, stream>>>(in_w, inw_h, EE * 2 * DIN * Cc);
    cast_f32_to_f16<<<(EE * 136 * DIN + TPB - 1) / TPB, TPB, 0, stream>>>(xproj_w, xpw_h, EE * 136 * DIN);
    cast_f32_to_f16<<<(EE * Cc * DIN + TPB - 1) / TPB, TPB, 0, stream>>>(out_w, outw_h, EE * Cc * DIN);
    cast_f32_to_f16<<<(Cc * Cc + TPB - 1) / TPB, TPB, 0, stream>>>(fusion_w, fusw_h, Cc * Cc);

    // 2) router
    router_conv_pool<<<Bn * CONVDIM, TPB, 0, stream>>>(x, r_conv_w, r_conv_b, pooled);
    router_softmax<<<1, 32, 0, stream>>>(pooled, r_lin_w, r_lin_b, wts);

    // 3) in-proj GEMM: XR(E,8192,512) = xseq(8192,128) @ in_w(E,512,128)^T
    {
        int Ntiles = (2 * DIN) / 16;                      // 32
        int stiles = (MROWS / 32) * ((Ntiles + 1) / 2);   // 4096
        dim3 g((stiles + 3) / 4, EE);
        gemm_wmma_f16<<<g, 128, 0, stream>>>(xseq_h, 0L,
                                             inw_h, (long)(2 * DIN) * Cc,
                                             XR, (long)MROWS * 2 * DIN,
                                             MROWS, Ntiles, 2 * DIN, Cc,
                                             Cc, Cc, 2 * DIN);
    }

    // 4) causal depthwise conv + SiLU
    dwconv_silu<<<(nEW + TPB - 1) / TPB, TPB, 0, stream>>>(XR, conv_w, conv_b, xsf, xsh);

    // 5) x-proj GEMM: dbl(E,8192,136) = xs(E,8192,256) @ xproj_w(E,136,256)^T
    {
        int Ntiles = 9;
        int stiles = (MROWS / 32) * ((Ntiles + 1) / 2);   // 1280
        dim3 g((stiles + 3) / 4, EE);
        gemm_wmma_f16<<<g, 128, 0, stream>>>(xsh, (long)MROWS * DIN,
                                             xpw_h, (long)136 * DIN,
                                             dbl, (long)MROWS * 136,
                                             MROWS, Ntiles, 136, DIN,
                                             DIN, DIN, 136);
    }

    // 6) delta = softplus(dt @ dt_w^T + dt_b)
    delta_kernel<<<(nEW + TPB - 1) / TPB, TPB, 0, stream>>>(dbl, dt_w, dt_b, delta);

    // 7) sequential scan (ys overwrites delta buffer)
    scan_kernel<<<EE * Bn, 256, 0, stream>>>(dbl, delta, xsf, A_log);

    // 8) gating: y = (ys + xs*D) * silu(res) -> f16 (reuse xsh buffer)
    ymix_kernel<<<(nEW + TPB - 1) / TPB, TPB, 0, stream>>>(delta, xsf, XR, D_p, xsh);

    // 9) out-proj GEMM: outE(E,8192,128) = y(E,8192,256) @ out_w(E,128,256)^T
    {
        int Ntiles = Cc / 16;                             // 8
        int stiles = (MROWS / 32) * ((Ntiles + 1) / 2);   // 1024
        dim3 g((stiles + 3) / 4, EE);
        gemm_wmma_f16<<<g, 128, 0, stream>>>(xsh, (long)MROWS * DIN,
                                             outw_h, (long)Cc * DIN,
                                             outE, (long)MROWS * Cc,
                                             MROWS, Ntiles, Cc, DIN,
                                             DIN, DIN, Cc);
    }

    // 10) expert mixture -> (B*HW, C) f16
    combine_kernel<<<(Bn * LL * Cc + TPB - 1) / TPB, TPB, 0, stream>>>(outE, wts, wsmix);

    // 11) fusion GEMM with transposed store + bias -> d_out (B,C,H,W) f32
    {
        int Ntiles = Cc / 16;                             // 8
        int tiles = (MROWS / 16) * Ntiles;                // 4096
        gemm_wmma_fusion<<<(tiles + 3) / 4, 128, 0, stream>>>(wsmix, fusw_h, fusion_b, out, Ntiles);
    }
}
